// HGAT_65678639891197
// MI455X (gfx1250) — matrix operand order repaired and checked by reading, not compiled, not run on hardware
//
#include <hip/hip_runtime.h>
#include <hip/hip_bf16.h>
#include <stdint.h>
#include <stddef.h>

// ---------------------------------------------------------------------------
// HGAT forward for MI455X (gfx1250, wave32).
// Dense 128x128 projections via v_wmma_f32_16x16x32_f16 (f32 accumulate);
// irregular GATv2 edge passes via f32 scalar + L2 atomics.
// ---------------------------------------------------------------------------

#define NWORDS   50000
#define NSENT    5000
#define NTOT     55000
#define CH       128
#define HEADS    8
#define HC       16
#define E_W2W    600000
#define E_W2S    100000
#define E_S2S    100000
#define E_SIM    100000

typedef __attribute__((ext_vector_type(16))) _Float16 v16h;
typedef __attribute__((ext_vector_type(8)))  float    v8f;

static inline int cdiv(int a, int b) { return (a + b - 1) / b; }

// ---- monotone float <-> uint encoding for atomicMax on floats --------------
__device__ __forceinline__ unsigned int enc_f32(float f) {
  unsigned int u = __float_as_uint(f);
  return (u & 0x80000000u) ? ~u : (u | 0x80000000u);
}
__device__ __forceinline__ float dec_f32(unsigned int u) {
  return (u & 0x80000000u) ? __uint_as_float(u & 0x7FFFFFFFu)
                           : __uint_as_float(~u);
}

// ---------------------------------------------------------------------------
// zero fill
__global__ void k_zero_f32(float* p, int n) {
  int t = blockIdx.x * blockDim.x + threadIdx.x;
  if (t < n) p[t] = 0.0f;
}

// f32 -> f16 convert
__global__ void k_f2h(const float* __restrict__ x, _Float16* __restrict__ y, int n) {
  int t = blockIdx.x * blockDim.x + threadIdx.x;
  if (t < n) y[t] = (_Float16)x[t];
}

// ---------------------------------------------------------------------------
// Pack a [128(K) x 128(N)] f32 weight into WMMA B-operand f16 layout.
// dst[((tn*4+kk)*32 + lane)*16 + i] = W[k, n], with
//   n = tn*16 + (lane&15),  k = kk*32 + (lane>=16 ? 8 : 0) + (i<8 ? i : i+8)
__global__ void k_pack_w(const float* __restrict__ W, _Float16* __restrict__ dst) {
  int t = blockIdx.x * 256 + threadIdx.x;     // 16384 threads
  if (t >= 16384) return;
  int i    = t & 15;
  int lane = (t >> 4) & 31;
  int tile = t >> 9;                          // 0..31
  int tn = tile >> 2, kk = tile & 3;
  int n = tn * 16 + (lane & 15);
  int k = kk * 32 + ((lane >= 16) ? 8 : 0) + ((i < 8) ? i : (i + 8));
  dst[t] = (_Float16)W[k * CH + n];
}

// ---------------------------------------------------------------------------
// Load one 16x32 f16 A tile (rows rbase..rbase+15, K = kk*32..) per WMMA layout.
__device__ __forceinline__ v16h load_a_tile(const _Float16* __restrict__ xh,
                                            int row, int kk, int hiA) {
  union { unsigned int u[8]; v16h h; } tmp;
  const unsigned int* p = (const unsigned int*)(xh + (size_t)row * CH) + kk * 16;
#pragma unroll
  for (int j = 0; j < 8; ++j) {
    int idx = ((j < 4) ? j : (j + 4)) + hiA;           // K pairs {0..7,16..23}(+8)
    tmp.u[j] = p[idx];
  }
  return tmp.h;
}

// ---------------------------------------------------------------------------
// GEMM: out[N,128] = xh[N,128](f16) @ Wpk(f16 packed), f32 accumulate.
// 8 waves / block; wave w computes rows [32*(blk*8+w), +32)  (2 M-tiles,
// shared B fetch -> 2 WMMA per ds-load pair, 64 WMMA per wave).
__global__ __launch_bounds__(256)
void k_gemm128_wmma(const _Float16* __restrict__ xh,
                    const _Float16* __restrict__ wpk,
                    float* __restrict__ out, int N) {
  __shared__ __align__(32) _Float16 ldsW[16384];      // 32 KB packed weights
  {
    const uint4* src = (const uint4*)wpk;
    uint4*       dst = (uint4*)ldsW;
    for (int i = threadIdx.x; i < 2048; i += 256) dst[i] = src[i];
  }
  __syncthreads();

  const int lane = threadIdx.x & 31;
  // rowbase is wave-uniform: force it into an SGPR so tile guards lower to
  // scalar branches instead of exec-mask save/restore.
  const int rowbase =
      __builtin_amdgcn_readfirstlane((blockIdx.x * 8 + (threadIdx.x >> 5)) * 32);
  if (rowbase >= N) return;                            // s_cmp + s_cbranch

  int r0 = rowbase + (lane & 15);
  int r1 = rowbase + 16 + (lane & 15);
  if (r0 >= N) r0 = N - 1;                             // clamp (EXEC stays full)
  if (r1 >= N) r1 = N - 1;

  const int hiA = (lane >= 16) ? 4 : 0;                // +8 halves for hi half-wave
  v16h A0[4], A1[4];
#pragma unroll
  for (int kk = 0; kk < 4; ++kk) {
    A0[kk] = load_a_tile(xh, r0, kk, hiA);
    A1[kk] = load_a_tile(xh, r1, kk, hiA);
  }

  const bool fullTile = (rowbase + 32) <= N;           // scalar condition
  const int  mhi = (lane >= 16) ? 8 : 0;
  const int  col = (lane & 15);
  // One per-lane base pointer; every store below uses a compile-time offset
  // (t*64 + r*512 bytes <= 4032, fits the 24-bit immediate).
  float* __restrict__ b0 = out + (size_t)(rowbase + mhi) * CH + col;
  float* __restrict__ b1 = b0 + (size_t)16 * CH;

#pragma unroll
  for (int t = 0; t < 8; ++t) {                        // 8 column tiles of 16
    v8f c0 = {}, c1 = {};
#pragma unroll
    for (int kk = 0; kk < 4; ++kk) {
      const v16h* bp = (const v16h*)&ldsW[(((t * 4 + kk) * 32) + lane) * 16];
      v16h b = *bp;
      c0 = __builtin_amdgcn_wmma_f32_16x16x32_f16(false, A0[kk], false, b,
                                                  (short)0, c0, false, false);
      c1 = __builtin_amdgcn_wmma_f32_16x16x32_f16(false, A1[kk], false, b,
                                                  (short)0, c1, false, false);
    }
    if (fullTile) {                                    // uniform fast path
#pragma unroll
      for (int r = 0; r < 8; ++r) b0[t * 16 + r * CH] = c0[r];
#pragma unroll
      for (int r = 0; r < 8; ++r) b1[t * 16 + r * CH] = c1[r];
    } else {                                           // remainder wave only
#pragma unroll
      for (int r = 0; r < 8; ++r) {
        if (rowbase + r + mhi < N) b0[t * 16 + r * CH] = c0[r];
      }
#pragma unroll
      for (int r = 0; r < 8; ++r) {
        if (rowbase + 16 + r + mhi < N) b1[t * 16 + r * CH] = c1[r];
      }
    }
  }
}

// ---------------------------------------------------------------------------
// Edge pass 1: alpha[e,h] = dot(leaky(xl[src]+xr[dst](+ea@We)), att[h]);
//              amax[dst,h] = max(...)  (encoded uint)
__global__ void k_edge_alpha(const float* __restrict__ xl, const float* __restrict__ xr,
                             const int* __restrict__ src, const int* __restrict__ dst,
                             const float* __restrict__ ea, const float* __restrict__ We,
                             const float* __restrict__ att,
                             float* __restrict__ alpha, unsigned int* __restrict__ amax,
                             int E) {
  int t = blockIdx.x * blockDim.x + threadIdx.x;
  if (t >= E * HEADS) return;
  int e = t >> 3, h = t & 7;
  int s = src[e], d = dst[e];
  const float* pl = xl + (size_t)s * CH + h * HC;
  const float* pr = xr + (size_t)d * CH + h * HC;

  float medge[HC];
#pragma unroll
  for (int j = 0; j < HC; ++j) medge[j] = 0.0f;
  if (We != nullptr) {
    float ev[17];
#pragma unroll
    for (int q = 0; q < 17; ++q) ev[q] = ea[(size_t)e * 17 + q];
#pragma unroll
    for (int j = 0; j < HC; ++j) {
      float s17 = 0.0f;
#pragma unroll
      for (int q = 0; q < 17; ++q) s17 += ev[q] * We[q * CH + h * HC + j];
      medge[j] = s17;
    }
  }

  float acc = 0.0f;
#pragma unroll
  for (int j = 0; j < HC; ++j) {
    float m = pl[j] + pr[j] + medge[j];
    m = (m > 0.0f) ? m : 0.2f * m;                     // leaky_relu(0.2)
    acc += m * att[h * HC + j];
  }
  alpha[t] = acc;
  atomicMax(&amax[(size_t)d * HEADS + h], enc_f32(acc));
}

// Edge pass 2: e = exp(alpha - amax[dst]); denom[dst,h] += e
__global__ void k_edge_expsum(const int* __restrict__ dst,
                              const unsigned int* __restrict__ amax,
                              float* __restrict__ alpha, float* __restrict__ denom,
                              int E) {
  int t = blockIdx.x * blockDim.x + threadIdx.x;
  if (t >= E * HEADS) return;
  int e = t >> 3, h = t & 7;
  int d = dst[e];
  unsigned int em = amax[(size_t)d * HEADS + h];
  float mx = (em == 0u) ? 0.0f : dec_f32(em);          // 0 sentinel == -inf -> 0
  float ev = __expf(alpha[t] - mx);
  alpha[t] = ev;                                       // overwrite with exp value
  atomicAdd(&denom[(size_t)d * HEADS + h], ev);
}

// Edge pass 3: accum[dst, ch] += xl[src, ch] * e/(denom+1e-16)
__global__ void k_edge_scatter(const float* __restrict__ xl,
                               const int* __restrict__ src, const int* __restrict__ dst,
                               const float* __restrict__ alpha,
                               const float* __restrict__ denom,
                               float* __restrict__ accum, int E) {
  int t = blockIdx.x * blockDim.x + threadIdx.x;
  if (t >= E * CH) return;
  int e  = t >> 7;
  int ch = t & 127;
  int h  = ch >> 4;
  int s = src[e], d = dst[e];
  float ev  = alpha[(size_t)e * HEADS + h];
  float den = denom[(size_t)d * HEADS + h];
  float a = ev / (den + 1e-16f);
  atomicAdd(&accum[(size_t)d * CH + ch], xl[(size_t)s * CH + ch] * a);
}

// ---------------------------------------------------------------------------
// Epilogue: y = relu(layernorm(accum + bias)); write f16 (next GEMM) + f32.
// One wave per node (4 channels / lane).
__global__ __launch_bounds__(256)
void k_epilogue_ln(const float* __restrict__ accum, const float* __restrict__ bias,
                   const float* __restrict__ g, const float* __restrict__ b,
                   _Float16* __restrict__ xh, float* __restrict__ xf, int N) {
  int wave = threadIdx.x >> 5, lane = threadIdx.x & 31;
  int n = blockIdx.x * 8 + wave;
  if (n >= N) return;
  const float* row = accum + (size_t)n * CH;
  float v[4];
  float s = 0.0f;
#pragma unroll
  for (int i = 0; i < 4; ++i) {
    int ch = i * 32 + lane;
    v[i] = row[ch] + bias[ch];
    s += v[i];
  }
#pragma unroll
  for (int m = 16; m >= 1; m >>= 1) s += __shfl_xor(s, m, 32);
  float mu = s * (1.0f / 128.0f);
  float q = 0.0f;
#pragma unroll
  for (int i = 0; i < 4; ++i) { float d = v[i] - mu; q += d * d; }
#pragma unroll
  for (int m = 16; m >= 1; m >>= 1) q += __shfl_xor(q, m, 32);
  float rstd = rsqrtf(q * (1.0f / 128.0f) + 1e-5f);
#pragma unroll
  for (int i = 0; i < 4; ++i) {
    int ch = i * 32 + lane;
    float y = (v[i] - mu) * rstd * g[ch] + b[ch];
    y = fmaxf(y, 0.0f);
    xh[(size_t)n * CH + ch] = (_Float16)y;
    xf[(size_t)n * CH + ch] = y;
  }
}

// ---------------------------------------------------------------------------
// Classifier: out = relu(LN(x@W1+b1)) @ W2 + b2.  One wave per node.
__global__ __launch_bounds__(256)
void k_classifier(const float* __restrict__ xf,
                  const float* __restrict__ W1, const float* __restrict__ b1,
                  const float* __restrict__ lng, const float* __restrict__ lnb,
                  const float* __restrict__ W2, const float* __restrict__ b2,
                  float* __restrict__ out, int N) {
  int wave = threadIdx.x >> 5, lane = threadIdx.x & 31;
  int n = blockIdx.x * 8 + wave;
  if (n >= N) return;
  const float* row = xf + (size_t)n * CH;
  float h = b1[lane];
  for (int k = 0; k < CH; ++k) h += row[k] * W1[k * 32 + lane];
  float s = h;
#pragma unroll
  for (int m = 16; m >= 1; m >>= 1) s += __shfl_xor(s, m, 32);
  float mu = s * (1.0f / 32.0f);
  float d = h - mu;
  float q = d * d;
#pragma unroll
  for (int m = 16; m >= 1; m >>= 1) q += __shfl_xor(q, m, 32);
  float rstd = rsqrtf(q * (1.0f / 32.0f) + 1e-5f);
  float y = d * rstd * lng[lane] + lnb[lane];
  y = fmaxf(y, 0.0f);
  float p0 = y * W2[lane * 2 + 0];
  float p1 = y * W2[lane * 2 + 1];
#pragma unroll
  for (int m = 16; m >= 1; m >>= 1) {
    p0 += __shfl_xor(p0, m, 32);
    p1 += __shfl_xor(p1, m, 32);
  }
  if (lane == 0) {
    out[(size_t)n * 2 + 0] = p0 + b2[0];
    out[(size_t)n * 2 + 1] = p1 + b2[1];
  }
}

// ---------------------------------------------------------------------------
// Host-side GAT layer driver (all launches on `stream`).
static void run_gat(hipStream_t stream,
                    const _Float16* xh_in, int N,
                    const int* ei, int E,
                    const _Float16* wl_pk, const _Float16* wr_pk,
                    const float* att, const float* bias,
                    const float* ln_g, const float* ln_b,
                    const float* ea, const float* We,
                    float* xl, float* xr, float* accum, float* alpha,
                    unsigned int* amax, float* denom,
                    _Float16* xh_out, float* xf_out) {
  const int* src = ei;
  const int* dst = ei + E;

  k_zero_f32<<<cdiv(N * CH, 256), 256, 0, stream>>>(accum, N * CH);
  k_zero_f32<<<cdiv(N * HEADS, 256), 256, 0, stream>>>((float*)amax, N * HEADS);
  k_zero_f32<<<cdiv(N * HEADS, 256), 256, 0, stream>>>(denom, N * HEADS);

  int gemm_blocks = cdiv(cdiv(N, 32), 8);
  k_gemm128_wmma<<<gemm_blocks, 256, 0, stream>>>(xh_in, wl_pk, xl, N);
  k_gemm128_wmma<<<gemm_blocks, 256, 0, stream>>>(xh_in, wr_pk, xr, N);

  k_edge_alpha<<<cdiv(E * HEADS, 256), 256, 0, stream>>>(xl, xr, src, dst, ea, We,
                                                         att, alpha, amax, E);
  k_edge_expsum<<<cdiv(E * HEADS, 256), 256, 0, stream>>>(dst, amax, alpha, denom, E);
  k_edge_scatter<<<cdiv(E * CH, 256), 256, 0, stream>>>(xl, src, dst, alpha, denom,
                                                        accum, E);
  k_epilogue_ln<<<cdiv(N, 8), 256, 0, stream>>>(accum, bias, ln_g, ln_b,
                                                xh_out, xf_out, N);
}

// ---------------------------------------------------------------------------
extern "C" void kernel_launch(void* const* d_in, const int* in_sizes, int n_in,
                              void* d_out, int out_size, void* d_ws, size_t ws_size,
                              hipStream_t stream) {
  (void)in_sizes; (void)n_in; (void)out_size; (void)ws_size;

  // ---- input mapping (setup_inputs order; params pytree sorted-key order) --
  const float* x_words  = (const float*)d_in[0];
  const float* x_sent   = (const float*)d_in[1];
  const int*   w2w_ei   = (const int*)d_in[2];
  const int*   w2s_ei   = (const int*)d_in[3];
  const int*   s2s_ei   = (const int*)d_in[4];
  const float* s2s_type = (const float*)d_in[5];
  const int*   sim_ei   = (const int*)d_in[6];
  // cls: W1, W2, b1, b2, lnb, lng
  const float* cls_W1  = (const float*)d_in[7];
  const float* cls_W2  = (const float*)d_in[8];
  const float* cls_b1  = (const float*)d_in[9];
  const float* cls_b2  = (const float*)d_in[10];
  const float* cls_lnb = (const float*)d_in[11];
  const float* cls_lng = (const float*)d_in[12];
  // ln1..ln7: (b, g) each
  const float* ln_b_[7], *ln_g_[7];
  for (int i = 0; i < 7; ++i) {
    ln_b_[i] = (const float*)d_in[13 + 2 * i];
    ln_g_[i] = (const float*)d_in[14 + 2 * i];
  }
  // red1/red2: Wl, Wr, att, b
  const float* red1_Wl = (const float*)d_in[27]; const float* red1_Wr = (const float*)d_in[28];
  const float* red1_at = (const float*)d_in[29]; const float* red1_b  = (const float*)d_in[30];
  const float* red2_Wl = (const float*)d_in[31]; const float* red2_Wr = (const float*)d_in[32];
  const float* red2_at = (const float*)d_in[33]; const float* red2_b  = (const float*)d_in[34];
  // s2s1/s2s2: We, Wl, Wr, att, b
  const float* s2s1_We = (const float*)d_in[35];
  const float* s2s1_Wl = (const float*)d_in[36]; const float* s2s1_Wr = (const float*)d_in[37];
  const float* s2s1_at = (const float*)d_in[38]; const float* s2s1_b  = (const float*)d_in[39];
  const float* s2s2_We = (const float*)d_in[40];
  const float* s2s2_Wl = (const float*)d_in[41]; const float* s2s2_Wr = (const float*)d_in[42];
  const float* s2s2_at = (const float*)d_in[43]; const float* s2s2_b  = (const float*)d_in[44];
  // w2s / w2w1: Wl, Wr, att, b
  const float* w2s_Wl  = (const float*)d_in[45]; const float* w2s_Wr  = (const float*)d_in[46];
  const float* w2s_at  = (const float*)d_in[47]; const float* w2s_b   = (const float*)d_in[48];
  const float* w2w_Wl  = (const float*)d_in[49]; const float* w2w_Wr  = (const float*)d_in[50];
  const float* w2w_at  = (const float*)d_in[51]; const float* w2w_b   = (const float*)d_in[52];

  float* out = (float*)d_out;

  // ---- workspace carve-up --------------------------------------------------
  uint8_t* ws = (uint8_t*)d_ws;
  size_t off = 0;
  auto alloc = [&](size_t bytes) -> void* {
    void* p = ws + off;
    off += (bytes + 255) & ~(size_t)255;
    return p;
  };
  _Float16* xh    = (_Float16*)alloc((size_t)NTOT * CH * 2);
  float*    xl    = (float*)alloc((size_t)NTOT * CH * 4);
  float*    xr    = (float*)alloc((size_t)NTOT * CH * 4);
  float*    accum = (float*)alloc((size_t)NTOT * CH * 4);
  float*    xf    = (float*)alloc((size_t)NTOT * CH * 4);
  float*    alpha = (float*)alloc((size_t)E_W2W * HEADS * 4);
  unsigned int* amax = (unsigned int*)alloc((size_t)NTOT * HEADS * 4);
  float*    denom = (float*)alloc((size_t)NTOT * HEADS * 4);
  _Float16* wpk   = (_Float16*)alloc((size_t)12 * 16384 * 2);

  _Float16* pk_w2w_Wl  = wpk + 0 * 16384; _Float16* pk_w2w_Wr  = wpk + 1 * 16384;
  _Float16* pk_w2s_Wl  = wpk + 2 * 16384; _Float16* pk_w2s_Wr  = wpk + 3 * 16384;
  _Float16* pk_s2s1_Wl = wpk + 4 * 16384; _Float16* pk_s2s1_Wr = wpk + 5 * 16384;
  _Float16* pk_s2s2_Wl = wpk + 6 * 16384; _Float16* pk_s2s2_Wr = wpk + 7 * 16384;
  _Float16* pk_red1_Wl = wpk + 8 * 16384; _Float16* pk_red1_Wr = wpk + 9 * 16384;
  _Float16* pk_red2_Wl = wpk + 10 * 16384; _Float16* pk_red2_Wr = wpk + 11 * 16384;

  // ---- pack all dense weights into WMMA B layout ---------------------------
  k_pack_w<<<64, 256, 0, stream>>>(w2w_Wl,  pk_w2w_Wl);
  k_pack_w<<<64, 256, 0, stream>>>(w2w_Wr,  pk_w2w_Wr);
  k_pack_w<<<64, 256, 0, stream>>>(w2s_Wl,  pk_w2s_Wl);
  k_pack_w<<<64, 256, 0, stream>>>(w2s_Wr,  pk_w2s_Wr);
  k_pack_w<<<64, 256, 0, stream>>>(s2s1_Wl, pk_s2s1_Wl);
  k_pack_w<<<64, 256, 0, stream>>>(s2s1_Wr, pk_s2s1_Wr);
  k_pack_w<<<64, 256, 0, stream>>>(s2s2_Wl, pk_s2s2_Wl);
  k_pack_w<<<64, 256, 0, stream>>>(s2s2_Wr, pk_s2s2_Wr);
  k_pack_w<<<64, 256, 0, stream>>>(red1_Wl, pk_red1_Wl);
  k_pack_w<<<64, 256, 0, stream>>>(red1_Wr, pk_red1_Wr);
  k_pack_w<<<64, 256, 0, stream>>>(red2_Wl, pk_red2_Wl);
  k_pack_w<<<64, 256, 0, stream>>>(red2_Wr, pk_red2_Wr);

  // ---- initial activations -> f16 -----------------------------------------
  k_f2h<<<cdiv(NWORDS * CH, 256), 256, 0, stream>>>(x_words, xh, NWORDS * CH);
  k_f2h<<<cdiv(NSENT * CH, 256), 256, 0, stream>>>(x_sent, xh + (size_t)NWORDS * CH,
                                                   NSENT * CH);

  // L1: words, w2w edges, params w2w1, ln1  (writes xh[0:NW])
  run_gat(stream, xh, NWORDS, w2w_ei, E_W2W, pk_w2w_Wl, pk_w2w_Wr,
          w2w_at, w2w_b, ln_g_[0], ln_b_[0], nullptr, nullptr,
          xl, xr, accum, alpha, amax, denom, xh, xf);
  // L2: same graph/params, ln2
  run_gat(stream, xh, NWORDS, w2w_ei, E_W2W, pk_w2w_Wl, pk_w2w_Wr,
          w2w_at, w2w_b, ln_g_[1], ln_b_[1], nullptr, nullptr,
          xl, xr, accum, alpha, amax, denom, xh, xf);
  // L3: concat(words, sent) = full xh, w2s edges, ln3 (writes all NTOT rows)
  run_gat(stream, xh, NTOT, w2s_ei, E_W2S, pk_w2s_Wl, pk_w2s_Wr,
          w2s_at, w2s_b, ln_g_[2], ln_b_[2], nullptr, nullptr,
          xl, xr, accum, alpha, amax, denom, xh, xf);
  // L4: xs = rows [NW:], s2s edges with edge attrs, s2s1, ln4 -> xh[0:NS]
  run_gat(stream, xh + (size_t)NWORDS * CH, NSENT, s2s_ei, E_S2S,
          pk_s2s1_Wl, pk_s2s1_Wr, s2s1_at, s2s1_b, ln_g_[3], ln_b_[3],
          s2s_type, s2s1_We,
          xl, xr, accum, alpha, amax, denom, xh, xf);
  // L5: s2s2, ln5
  run_gat(stream, xh, NSENT, s2s_ei, E_S2S, pk_s2s2_Wl, pk_s2s2_Wr,
          s2s2_at, s2s2_b, ln_g_[4], ln_b_[4], s2s_type, s2s2_We,
          xl, xr, accum, alpha, amax, denom, xh, xf);
  // L6: sim graph, red1, ln6
  run_gat(stream, xh, NSENT, sim_ei, E_SIM, pk_red1_Wl, pk_red1_Wr,
          red1_at, red1_b, ln_g_[5], ln_b_[5], nullptr, nullptr,
          xl, xr, accum, alpha, amax, denom, xh, xf);
  // L7: red2, ln7 (final xr in xf[0:NS])
  run_gat(stream, xh, NSENT, sim_ei, E_SIM, pk_red2_Wl, pk_red2_Wr,
          red2_at, red2_b, ln_g_[6], ln_b_[6], nullptr, nullptr,
          xl, xr, accum, alpha, amax, denom, xh, xf);

  // classifier
  k_classifier<<<cdiv(NSENT, 8), 256, 0, stream>>>(xf, cls_W1, cls_b1,
                                                   cls_lng, cls_lnb,
                                                   cls_W2, cls_b2, out, NSENT);
}
